// ScriptGreedyDecoder_5368709120421
// MI455X (gfx1250) — compile-verified, matrix operand order checked by reading, not run on hardware
//
#include <hip/hip_runtime.h>
#include <math.h>

// ---------------------------------------------------------------- constants
#define BB   16
#define TT   128
#define FIN  240
#define FINP 256              // K padded for WMMA
#define EDIM 1024
#define HH   320
#define VV   29
#define VVP  32               // N padded for WMMA
#define BLANK 28
#define JHD  512
#define KCAT (EDIM + HH)      // 1344
#define MAXS 3
#define BUF  (TT * MAXS)      // 384 labels per batch row
#define GATES (4 * HH)        // 1280
#define MROWS (BB * TT)       // 2048 encoder rows

// packed regions in d_ws (offsets in bf16 elements)
#define SZ_LSTM  ((GATES / 16) * (HH / 32) * 512)          // 409600
#define OFF_WIH0 0
#define OFF_WHH0 (OFF_WIH0 + SZ_LSTM)
#define OFF_WIH1 (OFF_WHH0 + SZ_LSTM)
#define OFF_WHH1 (OFF_WIH1 + SZ_LSTM)
#define OFF_WJ1  (OFF_WHH1 + SZ_LSTM)
#define SZ_WJ1   ((JHD / 16) * (KCAT / 32) * 512)          // 688128
#define OFF_WJ2  (OFF_WJ1 + SZ_WJ1)
#define SZ_WJ2   ((VVP / 16) * (JHD / 32) * 512)           // 16384
#define OFF_WENC (OFF_WJ2 + SZ_WJ2)
#define SZ_WENC  ((EDIM / 16) * (FINP / 32) * 512)         // 262144
#define OFF_EMB  (OFF_WENC + SZ_WENC)
#define SZ_EMB   ((VV - 1) * HH)                           // 8960
#define OFF_XPK  (((OFF_EMB + SZ_EMB) + 511) & ~511)
#define SZ_XPK   ((MROWS / 16) * (FINP / 32) * 512)        // 524288
#define OFF_ENCB (OFF_XPK + SZ_XPK)
#define SZ_ENCB  (BB * TT * EDIM)                          // 2097152
#define OFF_BS0  (OFF_ENCB + SZ_ENCB)                      // combined f32 biases
#define OFF_BS1  (OFF_BS0 + 2 * GATES)                     // (f32 = 2 bf16 elems)

typedef __attribute__((ext_vector_type(16))) __bf16 v16bf;
typedef __attribute__((ext_vector_type(8)))  __bf16 v8bf;
typedef __attribute__((ext_vector_type(8)))  float  v8f;
typedef __attribute__((ext_vector_type(4)))  int    v4i_t;

// gfx1250 async global->LDS path (ASYNCcnt), guarded so the file compiles
// whether or not this toolchain declares the builtin.
#if defined(__AMDGCN__) && __has_builtin(__builtin_amdgcn_global_load_async_to_lds_b128)
#define HAS_ASYNC_LDS 1
typedef __attribute__((address_space(1))) v4i_t* as1_v4i;
typedef __attribute__((address_space(3))) v4i_t* as3_v4i;
__device__ __forceinline__ void async_copy_b128(const void* g, void* l) {
    __builtin_amdgcn_global_load_async_to_lds_b128((as1_v4i)g, (as3_v4i)l, 0, 0);
}
__device__ __forceinline__ void wait_async0() {
#if __has_builtin(__builtin_amdgcn_s_wait_asynccnt)
    __builtin_amdgcn_s_wait_asynccnt(0);
#else
    asm volatile("s_wait_asynccnt 0x0" ::: "memory");
#endif
}
#endif

// K index within a 16-bit A/B fragment for VGPR pair v (0..7) and lane-half
__device__ __forceinline__ int frag_k(int v, int half) {
    return ((v >> 2) << 4) + (half << 3) + ((v & 3) << 1);
}

__device__ __forceinline__ v8f wmma_bf16(v16bf a, v16bf b, v8f c) {
    return __builtin_amdgcn_wmma_f32_16x16x32_bf16(
        /*neg_a=*/false, a, /*neg_b=*/false, b,
        /*c_mod=*/(short)0, c, /*reuse_a=*/false, /*reuse_b=*/false);
}

// A fragment (16x32 bf16) from row-major bf16 LDS, row stride lda (16B aligned):
// two contiguous 16-byte LDS loads per lane.
__device__ __forceinline__ v16bf load_a_lds(const __bf16* A, int lda, int k0) {
    const int lane = threadIdx.x & 31, m = lane & 15, half = lane >> 4;
    const __bf16* p = A + m * lda + k0 + half * 8;
    v8bf lo = *reinterpret_cast<const v8bf*>(p);
    v8bf hi = *reinterpret_cast<const v8bf*>(p + 16);
    return __builtin_shufflevector(lo, hi, 0, 1, 2, 3, 4, 5, 6, 7,
                                   8, 9, 10, 11, 12, 13, 14, 15);
}

// fragment from packed storage: one 32-byte contiguous load per lane
__device__ __forceinline__ v16bf load_frag_pk(const __bf16* P, int tile) {
    const int lane = threadIdx.x & 31;
    return *reinterpret_cast<const v16bf*>(P + (size_t)tile * 512 + lane * 16);
}

__device__ __forceinline__ float sigf(float x) { return 1.f / (1.f + __expf(-x)); }

// ---------------------------------------------------------------- packers
// logical B[k,n] = W[n*K + k] (W row-major [N,K]); N%16==0, K%32==0
__global__ void pack_b_NK(const float* __restrict__ W, __bf16* __restrict__ P,
                          int N, int K) {
    int idx = blockIdx.x * blockDim.x + threadIdx.x;
    if (idx >= (N / 16) * (K / 32) * 512) return;
    int e = idx & 15, lane = (idx >> 4) & 31, tile = idx >> 9;
    int nks = K / 32, nt = tile / nks, ks = tile - nt * nks;
    int n = nt * 16 + (lane & 15);
    int k = ks * 32 + frag_k(e >> 1, lane >> 4) + (e & 1);
    P[idx] = (__bf16)W[(size_t)n * K + k];
}

// logical B[k,n] = W[k*N + n] (W row-major [K,N]); zero-padded to kpad x npad
__global__ void pack_b_KN(const float* __restrict__ W, __bf16* __restrict__ P,
                          int K, int N, int kpad, int npad) {
    int idx = blockIdx.x * blockDim.x + threadIdx.x;
    if (idx >= (npad / 16) * (kpad / 32) * 512) return;
    int e = idx & 15, lane = (idx >> 4) & 31, tile = idx >> 9;
    int nks = kpad / 32, nt = tile / nks, ks = tile - nt * nks;
    int n = nt * 16 + (lane & 15);
    int k = ks * 32 + frag_k(e >> 1, lane >> 4) + (e & 1);
    P[idx] = (__bf16)((k < K && n < N) ? W[(size_t)k * N + n] : 0.f);
}

// A-fragment pack of X row-major [M,K] (zero-padded K -> kpad); M%16==0
__global__ void pack_a_frag(const float* __restrict__ X, __bf16* __restrict__ P,
                            int M, int K, int kpad) {
    int idx = blockIdx.x * blockDim.x + threadIdx.x;
    if (idx >= (M / 16) * (kpad / 32) * 512) return;
    int e = idx & 15, lane = (idx >> 4) & 31, tile = idx >> 9;
    int nks = kpad / 32, tm = tile / nks, ks = tile - tm * nks;
    int m = tm * 16 + (lane & 15);
    int k = ks * 32 + frag_k(e >> 1, lane >> 4) + (e & 1);
    P[idx] = (__bf16)((k < K) ? X[(size_t)m * K + k] : 0.f);
}

__global__ void pack_f32_to_bf16(const float* __restrict__ W,
                                 __bf16* __restrict__ P, int n) {
    int idx = blockIdx.x * blockDim.x + threadIdx.x;
    if (idx < n) P[idx] = (__bf16)W[idx];
}

__global__ void add_bias_f32(const float* __restrict__ a,
                             const float* __restrict__ b,
                             float* __restrict__ o, int n) {
    int idx = blockIdx.x * blockDim.x + threadIdx.x;
    if (idx < n) o[idx] = a[idx] + b[idx];
}

// ---------------------------------------------------------------- encoder
// logits[2048,1024] = x @ W_enc + b; both operands pre-packed in fragment
// order -> inner loop is pure b128 loads + v_wmma. Also emits bf16 copy.
__global__ void __launch_bounds__(256)
encoder_gemm(const __bf16* __restrict__ xpk, const __bf16* __restrict__ Wpk,
             const float* __restrict__ bias, float* __restrict__ out,
             __bf16* __restrict__ outbf) {
    const int wave = threadIdx.x >> 5;
    const int tile = blockIdx.x * 8 + wave;          // 8192 tiles total
    const int tm = tile >> 6;                        // 128 M tiles
    const int tn = tile & 63;                        // 64  N tiles
    const int NKS = FINP / 32;                       // 8

    v8f acc = {};
#pragma unroll
    for (int ks = 0; ks < NKS; ++ks)
        acc = wmma_bf16(load_frag_pk(xpk, tm * NKS + ks),
                        load_frag_pk(Wpk, tn * NKS + ks), acc);

    const int lane = threadIdx.x & 31, half = lane >> 4;
    const int n = tn * 16 + (lane & 15);
    const float bv = bias[n];
#pragma unroll
    for (int r = 0; r < 8; ++r) {
        int m = tm * 16 + r + 8 * half;
        float v = acc[r] + bv;
        out[(size_t)m * EDIM + n] = v;
        outbf[(size_t)m * EDIM + n] = (__bf16)v;
    }
}

// ---------------------------------------------------------------- LSTM cell
__device__ void lstm_cell(const __bf16* X, const __bf16* Hp, const float* Cp,
                          const __bf16* pWih, const __bf16* pWhh,
                          const float* bsum,
                          __bf16* Hout, int ldh, float* Cout,
                          int wave, int lane) {
    const int NKS = HH / 32;                                // 10 k-steps
    for (int chunk = wave; chunk < HH / 16; chunk += 8) {   // wave-uniform
        v8f acc[4] = {};
#pragma unroll 2
        for (int ks = 0; ks < NKS; ++ks) {
            v16bf ax = load_a_lds(X,  HH, ks * 32);
            v16bf ah = load_a_lds(Hp, HH, ks * 32);
#pragma unroll
            for (int g = 0; g < 4; ++g) {
                const int tile = (g * 20 + chunk) * NKS + ks;
                acc[g] = wmma_bf16(ax, load_frag_pk(pWih, tile), acc[g]);
                acc[g] = wmma_bf16(ah, load_frag_pk(pWhh, tile), acc[g]);
            }
        }
        const int half = lane >> 4, col = chunk * 16 + (lane & 15);
#pragma unroll
        for (int r = 0; r < 8; ++r) {
            const int m = r + 8 * half;
            float gi = sigf (acc[0][r] + bsum[0*HH + col]);
            float gf = sigf (acc[1][r] + bsum[1*HH + col]);
            float gg = tanhf(acc[2][r] + bsum[2*HH + col]);
            float go = sigf (acc[3][r] + bsum[3*HH + col]);
            float c2 = gf * Cp[m * HH + col] + gi * gg;
            Hout[m * ldh + col] = (__bf16)(go * tanhf(c2));
            Cout[m * HH + col]  = c2;
        }
    }
}

// ---------------------------------------------------------------- decoder
__global__ void __launch_bounds__(256)
rnnt_decode(const __bf16* __restrict__ encbf,   // [16,128,1024] bf16
            const int*   __restrict__ lens,
            const __bf16* __restrict__ embbf,   // [28,320] bf16
            const __bf16* __restrict__ pWih0, const __bf16* __restrict__ pWhh0,
            const float* __restrict__ bsum0,
            const __bf16* __restrict__ pWih1, const __bf16* __restrict__ pWhh1,
            const float* __restrict__ bsum1,
            const __bf16* __restrict__ pWj1,  const float* __restrict__ bj1,
            const __bf16* __restrict__ pWj2,  const float* __restrict__ bj2,
            int* __restrict__ lens_out, int* __restrict__ labels_out,
            int* __restrict__ counts_out) {
    extern __shared__ char smem[];
    __bf16* Acat = (__bf16*)smem;               // [16][1344] joint A (enc | h1cur)
    __bf16* Jh   = Acat + 16 * KCAT;            // [16][512]  relu joint hidden
    __bf16* Ein  = Jh   + 16 * JHD;             // [16][320]  embedding
    __bf16* Hst0 = Ein  + 16 * HH;              // committed LSTM0 hidden
    __bf16* Hst1 = Hst0 + 16 * HH;              // committed LSTM1 hidden
    __bf16* Hc0  = Hst1 + 16 * HH;              // current-step LSTM0 hidden
    float*  Cst0 = (float*)(Hc0 + 16 * HH);     // committed cells
    float*  Cst1 = Cst0 + 16 * HH;
    float*  C0c  = Cst1 + 16 * HH;              // current-step cells
    float*  C1c  = C0c  + 16 * HH;
    float*  Lg   = C1c  + 16 * HH;              // [16][32] logits (pad 29->32)
    int*    last = (int*)(Lg + 16 * 32);
    int*    cnt  = last + 16;
    int*    nb   = cnt  + 16;
    int*    emit = nb   + 16;
    int*    Lb   = emit + 16;

    const int tid = threadIdx.x, lane = tid & 31, wave = tid >> 5;

    // warm L2 with the packed recurrent weights (global_prefetch_b8)
    for (int i = tid * 32; i < 4 * SZ_LSTM + SZ_WJ1; i += 256 * 32)
        __builtin_prefetch(pWih0 + i, 0, 1);

    for (int i = tid; i < 16 * HH; i += 256) {
        Hst0[i] = (__bf16)0.f; Hst1[i] = (__bf16)0.f;
        Cst0[i] = 0.f;         Cst1[i] = 0.f;
    }
    if (tid < 16) {
        last[tid] = -1;  cnt[tid] = 0;
        Lb[tid] = lens[tid];  lens_out[tid] = lens[tid];
    }
    for (int i = tid; i < BB * BUF; i += 256) labels_out[i] = -1;
    __syncthreads();

    for (int t = 0; t < TT; ++t) {
        // stage this timestep's features into Acat[:, :1024]
#ifdef HAS_ASYNC_LDS
        for (int tr = tid; tr < 16 * EDIM / 8; tr += 256) {   // 2048 x b128
            int m = tr >> 7, jc = tr & 127;
            async_copy_b128(encbf + ((size_t)m * TT + t) * EDIM + jc * 8,
                            Acat + m * KCAT + jc * 8);
        }
        wait_async0();
#else
        for (int i = tid; i < 16 * EDIM; i += 256) {
            int m = i >> 10, j = i & (EDIM - 1);
            Acat[m * KCAT + j] = encbf[((size_t)m * TT + t) * EDIM + j];
        }
#endif
        if (tid < 16) nb[tid] = 1;
        __syncthreads();

        for (int s = 0; s < MAXS; ++s) {
            // embedding of last symbol (SOS -> zeros)
            for (int i = tid; i < 16 * HH; i += 256) {
                int m = i / HH, j = i - m * HH;
                int l = last[m];
                int adj = (l > BLANK) ? l - 1 : l;
                adj = max(0, min(adj, VV - 2));
                Ein[i] = (l == -1) ? (__bf16)0.f : embbf[adj * HH + j];
            }
            __syncthreads();

            lstm_cell(Ein, Hst0, Cst0, pWih0, pWhh0, bsum0,
                      Hc0, HH, C0c, wave, lane);
            __syncthreads();
            // h1_cur lands in Acat[:,1024:1344] => joint A ready in place
            lstm_cell(Hc0, Hst1, Cst1, pWih1, pWhh1, bsum1,
                      Acat + EDIM, KCAT, C1c, wave, lane);
            __syncthreads();

            // joint layer 1: Jh = relu(Acat[16,1344] @ Wj1[1344,512] + bj1)
            for (int tn = wave; tn < JHD / 16; tn += 8) {       // wave-uniform
                v8f acc = {};
                for (int ks = 0; ks < KCAT / 32; ++ks) {
                    v16bf a = load_a_lds(Acat, KCAT, ks * 32);
                    acc = wmma_bf16(a, load_frag_pk(pWj1, tn * (KCAT / 32) + ks), acc);
                }
                const int half = lane >> 4, n = tn * 16 + (lane & 15);
                const float bv = bj1[n];
#pragma unroll
                for (int r = 0; r < 8; ++r)
                    Jh[(r + 8 * half) * JHD + n] = (__bf16)fmaxf(acc[r] + bv, 0.f);
            }
            __syncthreads();

            // joint layer 2: logits[16,29] = Jh @ Wj2[512,29] + bj2
            if (wave < 2) {                                     // wave-uniform
                v8f acc = {};
#pragma unroll 4
                for (int ks = 0; ks < JHD / 32; ++ks) {
                    v16bf a = load_a_lds(Jh, JHD, ks * 32);
                    acc = wmma_bf16(a, load_frag_pk(pWj2, wave * (JHD / 32) + ks), acc);
                }
                const int half = lane >> 4, n = wave * 16 + (lane & 15);
#pragma unroll
                for (int r = 0; r < 8; ++r) {
                    int m = r + 8 * half;
                    Lg[m * 32 + n] = (n < VV) ? acc[r] + bj2[n] : -1e30f;
                }
            }
            __syncthreads();

            // per-row argmax + predicated transducer state machine
            if (tid < 16) {
                const int m = tid;
                float best = Lg[m * 32]; int bk = 0;
                for (int k = 1; k < VV; ++k) {
                    float v = Lg[m * 32 + k];
                    if (v > best) { best = v; bk = k; }
                }
                const bool isb    = (bk == BLANK);
                const bool active = (nb[m] != 0) && (t < Lb[m]);
                const bool em     = active && !isb;
                if (em) {
                    labels_out[m * BUF + cnt[m]] = bk;
                    cnt[m] += 1;
                    last[m] = bk;
                }
                emit[m] = em ? 1 : 0;
                if (isb) nb[m] = 0;
            }
            __syncthreads();

            // commit hidden/cell state only for rows that emitted
            for (int i = tid; i < 16 * HH; i += 256) {
                int m = i / HH, j = i - m * HH;
                if (emit[m]) {
                    Hst0[m * HH + j] = Hc0[m * HH + j];
                    Cst0[m * HH + j] = C0c[m * HH + j];
                    Hst1[m * HH + j] = Acat[m * KCAT + EDIM + j];
                    Cst1[m * HH + j] = C1c[m * HH + j];
                }
            }
            __syncthreads();
        }
    }
    if (tid < 16) counts_out[tid] = cnt[tid];
}

// ---------------------------------------------------------------- launch
extern "C" void kernel_launch(void* const* d_in, const int* in_sizes, int n_in,
                              void* d_out, int out_size, void* d_ws, size_t ws_size,
                              hipStream_t stream) {
    (void)in_sizes; (void)n_in; (void)out_size; (void)ws_size;
    const float* x     = (const float*)d_in[0];
    const int*   lens  = (const int*)  d_in[1];
    const float* W_enc = (const float*)d_in[2];
    const float* b_enc = (const float*)d_in[3];
    const float* embed = (const float*)d_in[4];
    const float* Wih0  = (const float*)d_in[5];
    const float* Whh0  = (const float*)d_in[6];
    const float* bih0  = (const float*)d_in[7];
    const float* bhh0  = (const float*)d_in[8];
    const float* Wih1  = (const float*)d_in[9];
    const float* Whh1  = (const float*)d_in[10];
    const float* bih1  = (const float*)d_in[11];
    const float* bhh1  = (const float*)d_in[12];
    const float* Wj1   = (const float*)d_in[13];
    const float* bj1   = (const float*)d_in[14];
    const float* Wj2   = (const float*)d_in[15];
    const float* bj2   = (const float*)d_in[16];

    float* out_logits = (float*)d_out;                         // [16,128,1024]
    int*   out_lens   = (int*)(out_logits + (size_t)BB * TT * EDIM);
    int*   out_labels = out_lens + BB;                         // [16,384]
    int*   out_counts = out_labels + BB * BUF;                 // [16]

    __bf16* ws = (__bf16*)d_ws;
    __bf16* pWih0 = ws + OFF_WIH0;
    __bf16* pWhh0 = ws + OFF_WHH0;
    __bf16* pWih1 = ws + OFF_WIH1;
    __bf16* pWhh1 = ws + OFF_WHH1;
    __bf16* pWj1  = ws + OFF_WJ1;
    __bf16* pWj2  = ws + OFF_WJ2;
    __bf16* pWenc = ws + OFF_WENC;
    __bf16* pEmb  = ws + OFF_EMB;
    __bf16* pX    = ws + OFF_XPK;
    __bf16* encBF = ws + OFF_ENCB;
    float*  bsum0 = (float*)(ws + OFF_BS0);
    float*  bsum1 = (float*)(ws + OFF_BS1);

    // ---- one-time repack: weights/activations into WMMA fragment order ----
    const int TPB = 256;
    pack_b_NK<<<dim3((SZ_LSTM + TPB - 1) / TPB), dim3(TPB), 0, stream>>>(
        Wih0, pWih0, GATES, HH);
    pack_b_NK<<<dim3((SZ_LSTM + TPB - 1) / TPB), dim3(TPB), 0, stream>>>(
        Whh0, pWhh0, GATES, HH);
    pack_b_NK<<<dim3((SZ_LSTM + TPB - 1) / TPB), dim3(TPB), 0, stream>>>(
        Wih1, pWih1, GATES, HH);
    pack_b_NK<<<dim3((SZ_LSTM + TPB - 1) / TPB), dim3(TPB), 0, stream>>>(
        Whh1, pWhh1, GATES, HH);
    pack_b_KN<<<dim3((SZ_WJ1 + TPB - 1) / TPB), dim3(TPB), 0, stream>>>(
        Wj1, pWj1, KCAT, JHD, KCAT, JHD);
    pack_b_KN<<<dim3((SZ_WJ2 + TPB - 1) / TPB), dim3(TPB), 0, stream>>>(
        Wj2, pWj2, JHD, VV, JHD, VVP);
    pack_b_KN<<<dim3((SZ_WENC + TPB - 1) / TPB), dim3(TPB), 0, stream>>>(
        W_enc, pWenc, FIN, EDIM, FINP, EDIM);
    pack_a_frag<<<dim3((SZ_XPK + TPB - 1) / TPB), dim3(TPB), 0, stream>>>(
        x, pX, MROWS, FIN, FINP);
    pack_f32_to_bf16<<<dim3((SZ_EMB + TPB - 1) / TPB), dim3(TPB), 0, stream>>>(
        embed, pEmb, SZ_EMB);
    add_bias_f32<<<dim3((GATES + TPB - 1) / TPB), dim3(TPB), 0, stream>>>(
        bih0, bhh0, bsum0, GATES);
    add_bias_f32<<<dim3((GATES + TPB - 1) / TPB), dim3(TPB), 0, stream>>>(
        bih1, bhh1, bsum1, GATES);

    // ---- encoder: 8192 16x16 tiles, 8 waves (tiles) per block ----
    encoder_gemm<<<dim3(1024), dim3(256), 0, stream>>>(
        pX, pWenc, b_enc, out_logits, encBF);

    // ---- decoder: 1 workgroup, 8 waves, ~180 KB dynamic LDS (<=320 KB/WGP) --
    const size_t smem =
        (size_t)16 * KCAT * 2 + 16 * JHD * 2 + 4 * (16 * HH * 2) +  // bf16 bufs
        4 * (16 * HH * 4) +                                          // f32 cells
        16 * 32 * 4 +                                                // logits
        5 * 16 * 4 + 64;                                             // ints + pad
    rnnt_decode<<<dim3(1), dim3(256), smem, stream>>>(
        encBF, lens, pEmb,
        pWih0, pWhh0, bsum0, pWih1, pWhh1, bsum1,
        pWj1, bj1, pWj2, bj2,
        out_lens, out_labels, out_counts);
}
